// GMNLayer_73031623901579
// MI455X (gfx1250) — matrix operand order repaired and checked by reading, not compiled, not run on hardware
//
#include <hip/hip_runtime.h>

typedef __attribute__((ext_vector_type(16))) __bf16 v16bf;
typedef __attribute__((ext_vector_type(8)))  __bf16 v8bf;
typedef __attribute__((ext_vector_type(8)))  float  v8f;

// ---- fragment helpers ------------------------------------------------------
static __device__ inline v8bf cvt8(const float* __restrict__ p) {
  v8bf o;
#pragma unroll
  for (int i = 0; i < 8; ++i) o[i] = (__bf16)p[i];
  return o;
}
static __device__ inline v8bf ld8(const __bf16* p) {
  return *reinterpret_cast<const v8bf*>(p);
}
static __device__ inline v16bf cat16(v8bf lo, v8bf hi) {
  return __builtin_shufflevector(lo, hi, 0,1,2,3,4,5,6,7,8,9,10,11,12,13,14,15);
}
// A fragment (16-bit, 16x32): lanes 0-15 row m, halves K=kb+0..7 then kb+16..23;
// lanes 16-31 same rows, K=kb+8..15 then kb+24..31.  off = hi?8:0, row pre-offset by kb.
static __device__ inline v16bf afrag_g(const float* __restrict__ row, int off) {
  return cat16(cvt8(row + off), cvt8(row + 16 + off));
}
static __device__ inline v16bf afrag_s(const __bf16* row, int off) {
  return cat16(ld8(row + off), ld8(row + 16 + off));
}
// B fragment (16-bit, 32x16): lane holds column n = t*16+(lane&15), K=kb+off..+7
// then kb+16+off..+7 with off = (lane<16 ? 0 : 8).  Weights stored transposed [n][Kpad].
static __device__ inline v16bf bfrag(const __bf16* p) {
  return cat16(ld8(p), ld8(p + 16));
}
static __device__ inline v8f wmma_bf16(v16bf a, v16bf b, v8f c) {
  return __builtin_amdgcn_wmma_f32_16x16x32_bf16(false, a, false, b, (short)0, c,
                                                 false, false);
}
static __device__ inline float relu_f(float v) { return fmaxf(v, 0.0f); }

// ---- workspace prep --------------------------------------------------------
__global__ void egnn_zero_kernel(float4* __restrict__ p, int n4) {
  int i = blockIdx.x * blockDim.x + threadIdx.x;
  if (i < n4) p[i] = make_float4(0.f, 0.f, 0.f, 0.f);
}

// dst[n][k] = (k < Ksrc) ? src[k][n] : 0   (bf16, [128][Kpad])
__global__ void egnn_wconv_kernel(const float* __restrict__ src,
                                  __bf16* __restrict__ dst, int Ksrc, int Kpad) {
  int idx = blockIdx.x * blockDim.x + threadIdx.x;
  if (idx >= 128 * Kpad) return;
  int n = idx / Kpad;
  int k = idx - n * Kpad;
  float v = (k < Ksrc) ? src[k * 128 + n] : 0.0f;
  dst[idx] = (__bf16)v;
}

// ---- edge pipeline: edge_mlp + coord_mlp + scatter -------------------------
__global__ __launch_bounds__(256) void egnn_edge_kernel(
    const float* __restrict__ h, const float* __restrict__ coord,
    const int* __restrict__ erow, const int* __restrict__ ecol,
    const __bf16* __restrict__ wtE1, const float* __restrict__ be1,
    const __bf16* __restrict__ wtE2, const float* __restrict__ be2,
    const __bf16* __restrict__ wtC1, const float* __restrict__ bc1,
    const float* __restrict__ wc2,
    float* __restrict__ agg, float* __restrict__ coordsum, int E) {
  __shared__ __bf16 hidLds[8][16 * 128];
  __shared__ float  phiLds[8][16];

  const int wave = threadIdx.x >> 5;   // wave32
  const int lane = threadIdx.x & 31;
  const int l15  = lane & 15;
  const bool hiH = lane >= 16;
  const int off  = hiH ? 8 : 0;    // A-frag half offset (halves)
  const int koff = hiH ? 8 : 0;    // B-frag K offset (halves)
  const int mAdd = hiH ? 8 : 0;    // C-layout row offset

  const int edgeBase = (blockIdx.x * 8 + wave) * 16;
  if (edgeBase >= E) return;

  const int eIdx = min(edgeBase + l15, E - 1);
  const int r = erow[eIdx];
  const int c = ecol[eIdx];

  const float cdx = coord[r * 3 + 0] - coord[c * 3 + 0];
  const float cdy = coord[r * 3 + 1] - coord[c * 3 + 1];
  const float cdz = coord[r * 3 + 2] - coord[c * 3 + 2];
  const float radial = cdx * cdx + cdy * cdy + cdz * cdz;

  __bf16* myHid = hidLds[wave];
  v8f acc[8];

  // ---- GEMM1: e_in[16,288] = [h[row]|h[col]|radial,0..] times We1^T --------
#pragma unroll
  for (int t = 0; t < 8; ++t)
#pragma unroll
    for (int i = 0; i < 8; ++i) acc[t][i] = 0.0f;

#pragma unroll
  for (int ks = 0; ks < 8; ++ks) {
    const int kb = ks * 32;
    const float* srcRow = (kb < 128) ? (h + (size_t)r * 128 + kb)
                                     : (h + (size_t)c * 128 + (kb - 128));
    v16bf a = afrag_g(srcRow, off);
#pragma unroll
    for (int t = 0; t < 8; ++t) {
      v16bf b = bfrag(wtE1 + (size_t)(t * 16 + l15) * 288 + kb + koff);
      acc[t] = wmma_bf16(a, b, acc[t]);
    }
  }
  {  // k-step 8: only k=256 (radial) is non-zero
    v16bf a;
#pragma unroll
    for (int i = 0; i < 16; ++i) a[i] = (__bf16)0.0f;
    if (!hiH) a[0] = (__bf16)radial;
#pragma unroll
    for (int t = 0; t < 8; ++t) {
      v16bf b = bfrag(wtE1 + (size_t)(t * 16 + l15) * 288 + 256 + koff);
      acc[t] = wmma_bf16(a, b, acc[t]);
    }
  }
  // bias + relu -> LDS tile (bf16), C layout: (m = rg+mAdd, n = t*16+l15)
#pragma unroll
  for (int t = 0; t < 8; ++t) {
    const float bias = be1[t * 16 + l15];
#pragma unroll
    for (int rg = 0; rg < 8; ++rg) {
      const int mm = rg + mAdd;
      myHid[mm * 128 + t * 16 + l15] = (__bf16)relu_f(acc[t][rg] + bias);
    }
  }

  // ---- GEMM2: hid x We2^T -> edge_feat ------------------------------------
#pragma unroll
  for (int t = 0; t < 8; ++t)
#pragma unroll
    for (int i = 0; i < 8; ++i) acc[t][i] = 0.0f;
#pragma unroll
  for (int ks = 0; ks < 4; ++ks) {
    v16bf a = afrag_s(myHid + l15 * 128 + ks * 32, off);
#pragma unroll
    for (int t = 0; t < 8; ++t) {
      v16bf b = bfrag(wtE2 + (size_t)(t * 16 + l15) * 128 + ks * 32 + koff);
      acc[t] = wmma_bf16(a, b, acc[t]);
    }
  }
  // row index of edge handled by (rg, half): lane rg+mAdd holds it
  int rowOf[8];
#pragma unroll
  for (int rg = 0; rg < 8; ++rg) rowOf[rg] = __shfl(r, rg + mAdd);

  // bias + relu; scatter-add into agg; re-stage as bf16 for coord MLP
#pragma unroll
  for (int t = 0; t < 8; ++t) {
    const float bias = be2[t * 16 + l15];
#pragma unroll
    for (int rg = 0; rg < 8; ++rg) {
      const int mm = rg + mAdd;
      float v = relu_f(acc[t][rg] + bias);
      if (edgeBase + mm < E)
        atomicAdd(agg + (size_t)rowOf[rg] * 128 + t * 16 + l15, v);
      myHid[mm * 128 + t * 16 + l15] = (__bf16)v;
    }
  }

  // ---- GEMM3: edge_feat x Wc1^T, then phi = relu(.) . Wc2 -----------------
#pragma unroll
  for (int t = 0; t < 8; ++t)
#pragma unroll
    for (int i = 0; i < 8; ++i) acc[t][i] = 0.0f;
#pragma unroll
  for (int ks = 0; ks < 4; ++ks) {
    v16bf a = afrag_s(myHid + l15 * 128 + ks * 32, off);
#pragma unroll
    for (int t = 0; t < 8; ++t) {
      v16bf b = bfrag(wtC1 + (size_t)(t * 16 + l15) * 128 + ks * 32 + koff);
      acc[t] = wmma_bf16(a, b, acc[t]);
    }
  }
  float p[8];
#pragma unroll
  for (int rg = 0; rg < 8; ++rg) p[rg] = 0.0f;
#pragma unroll
  for (int t = 0; t < 8; ++t) {
    const float bias = bc1[t * 16 + l15];
    const float w    = wc2[t * 16 + l15];
#pragma unroll
    for (int rg = 0; rg < 8; ++rg)
      p[rg] += relu_f(acc[t][rg] + bias) * w;
  }
  // reduce partial dot across each 16-lane half (rows rg / rg+8)
#pragma unroll
  for (int s = 8; s >= 1; s >>= 1)
#pragma unroll
    for (int rg = 0; rg < 8; ++rg) p[rg] += __shfl_xor(p[rg], s, 16);
  if (lane == 0 || lane == 16) {
#pragma unroll
    for (int rg = 0; rg < 8; ++rg) phiLds[wave][rg + mAdd] = p[rg];
  }
  // trans = clip(coord_diff * phi); scatter mean pieces (lanes 0-15, edge = l15)
  if (!hiH && (edgeBase + l15) < E) {
    const float phi = phiLds[wave][l15];
    const float tx = fminf(fmaxf(cdx * phi, -100.0f), 100.0f);
    const float ty = fminf(fmaxf(cdy * phi, -100.0f), 100.0f);
    const float tz = fminf(fmaxf(cdz * phi, -100.0f), 100.0f);
    atomicAdd(coordsum + (size_t)r * 4 + 0, tx);
    atomicAdd(coordsum + (size_t)r * 4 + 1, ty);
    atomicAdd(coordsum + (size_t)r * 4 + 2, tz);
    atomicAdd(coordsum + (size_t)r * 4 + 3, 1.0f);
  }
}

// ---- node pipeline: node_mlp + residual + coord mean -----------------------
__global__ __launch_bounds__(256) void egnn_node_kernel(
    const float* __restrict__ h, const float* __restrict__ others,
    const float* __restrict__ agg, const float* __restrict__ coord,
    const __bf16* __restrict__ wtN1, const float* __restrict__ bn1,
    const __bf16* __restrict__ wtN2, const float* __restrict__ bn2,
    const float* __restrict__ coordsum,
    float* __restrict__ hOut, float* __restrict__ coordOut, int N) {
  __shared__ __bf16 hidLds[8][16 * 128];

  const int wave = threadIdx.x >> 5;
  const int lane = threadIdx.x & 31;
  const int l15  = lane & 15;
  const bool hiH = lane >= 16;
  const int off  = hiH ? 8 : 0;
  const int koff = hiH ? 8 : 0;
  const int mAdd = hiH ? 8 : 0;

  const int nodeBase = (blockIdx.x * 8 + wave) * 16;
  if (nodeBase >= N) return;
  const int node = min(nodeBase + l15, N - 1);

  __bf16* myHid = hidLds[wave];
  v8f acc[8];
#pragma unroll
  for (int t = 0; t < 8; ++t)
#pragma unroll
    for (int i = 0; i < 8; ++i) acc[t][i] = 0.0f;

  // GEMM-N1: n_in[16,384] = [others | h | agg] times Wn1^T
#pragma unroll
  for (int ks = 0; ks < 12; ++ks) {
    const int kb = ks * 32;
    const float* srcRow =
        (kb < 128) ? (others + (size_t)node * 128 + kb)
      : (kb < 256) ? (h + (size_t)node * 128 + (kb - 128))
                   : (agg + (size_t)node * 128 + (kb - 256));
    v16bf a = afrag_g(srcRow, off);
#pragma unroll
    for (int t = 0; t < 8; ++t) {
      v16bf b = bfrag(wtN1 + (size_t)(t * 16 + l15) * 384 + kb + koff);
      acc[t] = wmma_bf16(a, b, acc[t]);
    }
  }
#pragma unroll
  for (int t = 0; t < 8; ++t) {
    const float bias = bn1[t * 16 + l15];
#pragma unroll
    for (int rg = 0; rg < 8; ++rg) {
      const int mm = rg + mAdd;
      myHid[mm * 128 + t * 16 + l15] = (__bf16)relu_f(acc[t][rg] + bias);
    }
  }

  // GEMM-N2 + residual
#pragma unroll
  for (int t = 0; t < 8; ++t)
#pragma unroll
    for (int i = 0; i < 8; ++i) acc[t][i] = 0.0f;
#pragma unroll
  for (int ks = 0; ks < 4; ++ks) {
    v16bf a = afrag_s(myHid + l15 * 128 + ks * 32, off);
#pragma unroll
    for (int t = 0; t < 8; ++t) {
      v16bf b = bfrag(wtN2 + (size_t)(t * 16 + l15) * 128 + ks * 32 + koff);
      acc[t] = wmma_bf16(a, b, acc[t]);
    }
  }
#pragma unroll
  for (int t = 0; t < 8; ++t) {
    const float bias = bn2[t * 16 + l15];
#pragma unroll
    for (int rg = 0; rg < 8; ++rg) {
      const int nm = nodeBase + rg + mAdd;
      if (nm < N) {
        const size_t ix = (size_t)nm * 128 + t * 16 + l15;
        hOut[ix] = h[ix] + acc[t][rg] + bias;
      }
    }
  }

  // coord_out = coord + sums / max(counts, 1)
  if (!hiH && (nodeBase + l15) < N) {
    const int nd = nodeBase + l15;
    const float cnt = fmaxf(coordsum[(size_t)nd * 4 + 3], 1.0f);
#pragma unroll
    for (int j = 0; j < 3; ++j)
      coordOut[(size_t)nd * 3 + j] =
          coord[(size_t)nd * 3 + j] + coordsum[(size_t)nd * 4 + j] / cnt;
  }
}

// ---- launcher --------------------------------------------------------------
extern "C" void kernel_launch(void* const* d_in, const int* in_sizes, int n_in,
                              void* d_out, int out_size, void* d_ws, size_t ws_size,
                              hipStream_t stream) {
  const float* h      = (const float*)d_in[0];
  const float* coord  = (const float*)d_in[1];
  const float* others = (const float*)d_in[2];
  const int*   erow   = (const int*)d_in[3];
  const int*   ecol   = (const int*)d_in[4];
  const float* We1 = (const float*)d_in[5];  const float* be1 = (const float*)d_in[6];
  const float* We2 = (const float*)d_in[7];  const float* be2 = (const float*)d_in[8];
  const float* Wn1 = (const float*)d_in[9];  const float* bn1 = (const float*)d_in[10];
  const float* Wn2 = (const float*)d_in[11]; const float* bn2 = (const float*)d_in[12];
  const float* Wc1 = (const float*)d_in[13]; const float* bc1 = (const float*)d_in[14];
  const float* Wc2 = (const float*)d_in[15];

  const int N = in_sizes[1] / 3;   // coord is [N,3]
  const int E = in_sizes[3];       // edge_row is [E]

  char* ws = (char*)d_ws;
  size_t o = 0;
  float*  agg      = (float*)(ws + o); o += (size_t)N * 128 * sizeof(float);
  float*  coordsum = (float*)(ws + o); o += (size_t)N * 4 * sizeof(float);
  __bf16* wtE1 = (__bf16*)(ws + o); o += (size_t)128 * 288 * sizeof(__bf16);
  __bf16* wtE2 = (__bf16*)(ws + o); o += (size_t)128 * 128 * sizeof(__bf16);
  __bf16* wtC1 = (__bf16*)(ws + o); o += (size_t)128 * 128 * sizeof(__bf16);
  __bf16* wtN1 = (__bf16*)(ws + o); o += (size_t)128 * 384 * sizeof(__bf16);
  __bf16* wtN2 = (__bf16*)(ws + o); o += (size_t)128 * 128 * sizeof(__bf16);

  float* hOut     = (float*)d_out;
  float* coordOut = hOut + (size_t)N * 128;

  {  // zero agg + coordsum (contiguous); overflow into weight region is rewritten below
    const int n4 = (N * 132 + 3) / 4;
    egnn_zero_kernel<<<(n4 + 255) / 256, 256, 0, stream>>>((float4*)agg, n4);
  }
  egnn_wconv_kernel<<<(128 * 288 + 255) / 256, 256, 0, stream>>>(We1, wtE1, 257, 288);
  egnn_wconv_kernel<<<(128 * 128 + 255) / 256, 256, 0, stream>>>(We2, wtE2, 128, 128);
  egnn_wconv_kernel<<<(128 * 128 + 255) / 256, 256, 0, stream>>>(Wc1, wtC1, 128, 128);
  egnn_wconv_kernel<<<(128 * 384 + 255) / 256, 256, 0, stream>>>(Wn1, wtN1, 384, 384);
  egnn_wconv_kernel<<<(128 * 128 + 255) / 256, 256, 0, stream>>>(Wn2, wtN2, 128, 128);

  egnn_edge_kernel<<<(E + 127) / 128, 256, 0, stream>>>(
      h, coord, erow, ecol, wtE1, be1, wtE2, be2, wtC1, bc1, Wc2,
      agg, coordsum, E);
  egnn_node_kernel<<<(N + 127) / 128, 256, 0, stream>>>(
      h, others, agg, coord, wtN1, bn1, wtN2, bn2, coordsum,
      hOut, coordOut, N);
}
